// improved_advanced_GCN_4329327034533
// MI455X (gfx1250) — compile-verified, matrix operand order checked by reading, not run on hardware
//
#include <hip/hip_runtime.h>

typedef __attribute__((ext_vector_type(16))) _Float16 v16h;
typedef __attribute__((ext_vector_type(8)))  _Float16 v8h;
typedef __attribute__((ext_vector_type(8)))  float    v8f;

#define HC    128
#define EPSV  1e-5f

// ---------------------------------------------------------------------------
// A-fragment loader: row-major fp32 source -> f16 fragment (16x32 K-chunk).
// ISA layout (16-bit A 16x32): lane<16: M=lane, K = koff+0..7 (v0-3) and
// koff+16..23 (v4-7) with koff=0 ; lane>=16: same M rows, koff=8.
// ---------------------------------------------------------------------------
__device__ __forceinline__ v16h load_frag_f32(const float* __restrict__ base,
                                              int row, int ld, int kc,
                                              int lane) {
  int koff = (lane >> 4) << 3;                 // 0 or 8
  const float* p = base + (size_t)row * ld + kc + koff;
  v16h f;
#pragma unroll
  for (int i = 0; i < 8; ++i) f[i]     = (_Float16)p[i];
#pragma unroll
  for (int i = 0; i < 8; ++i) f[i + 8] = (_Float16)p[i + 16];
  return f;
}

// B-fragment loader from pre-converted f16 weights: 2x 16-byte loads, no VALU.
// B = W^T with W row-major [Hout x K] -> row = output column, same K pattern.
__device__ __forceinline__ v16h load_frag_f16(const _Float16* __restrict__ base,
                                              int row, int ld, int kc, int lane) {
  int koff = (lane >> 4) << 3;
  const _Float16* p = base + (size_t)row * ld + kc + koff;
  v8h lo = *(const v8h*)p;
  v8h hi = *(const v8h*)(p + 16);
  return __builtin_shufflevector(lo, hi, 0, 1, 2, 3, 4, 5, 6, 7,
                                 8, 9, 10, 11, 12, 13, 14, 15);
}

// C = A[M x K] * W[Hout x K]^T (+bias), Hout = NT*16, K multiple of 32.
// 256 threads = 8 waves; each wave computes a 16-row x Hout strip.
// B (and A across K-chunks) are software double-buffered so global loads for
// the next fragment issue while the current WMMA executes (no false reg dep).
// Optional fused epilogue: out2[r,c] = C[r,c] * dinv[r]^2 (GCN self-loop term,
// saves a full re-read of C per layer).
template <int NT>
__global__ void wmma_gemm_k(const float* __restrict__ A,
                            const _Float16* __restrict__ Wh,
                            const float* __restrict__ bias,
                            float* __restrict__ C,
                            const float* __restrict__ dinv,
                            float* __restrict__ out2,
                            int M, int K) {
  int wave = threadIdx.x >> 5;
  int lane = threadIdx.x & 31;
  int m0 = (blockIdx.x * (blockDim.x >> 5) + wave) << 4;
  if (m0 >= M) return;                         // uniform per-wave exit
  int l15 = lane & 15;
  int arow = m0 + l15;
  if (arow > M - 1) arow = M - 1;              // clamp tail loads

  v8f zero = {};
  v8f acc[NT];
#pragma unroll
  for (int t = 0; t < NT; ++t) acc[t] = zero;

  v16h a = load_frag_f32(A, arow, K, 0, lane);
  for (int kc = 0; kc < K; kc += 32) {
    v16h anext = a;
    if (kc + 32 < K) anext = load_frag_f32(A, arow, K, kc + 32, lane);
    v16h b = load_frag_f16(Wh, l15, K, kc, lane);
#pragma unroll
    for (int t = 0; t < NT; ++t) {
      v16h bnext = b;
      if (t + 1 < NT) bnext = load_frag_f16(Wh, (t + 1) * 16 + l15, K, kc, lane);
      acc[t] = __builtin_amdgcn_wmma_f32_16x16x32_f16(
          false, a, false, b, (short)0, acc[t], false, false);
      b = bnext;
    }
    a = anext;
  }

  // C/D layout: VGPR r -> M = r (lanes 0-15) / r+8 (lanes 16-31), N = lane&15
  int mbase = m0 + ((lane >> 4) << 3);
  const int Hout = NT * 16;
  float bv[NT];
#pragma unroll
  for (int t = 0; t < NT; ++t) bv[t] = bias ? bias[t * 16 + l15] : 0.0f;
#pragma unroll
  for (int r = 0; r < 8; ++r) {
    int row = mbase + r;
    if (row >= M) continue;
    float dv2 = 0.0f;
    if (out2) { float dv = dinv[row]; dv2 = dv * dv; }
    size_t roff = (size_t)row * Hout;
#pragma unroll
    for (int t = 0; t < NT; ++t) {
      int col = t * 16 + l15;
      float val = acc[t][r] + bv[t];
      C[roff + col] = val;
      if (out2) out2[roff + col] = val * dv2;
    }
  }
}

// ---------------------------------------------------------------------------
// Elementwise / scatter helpers
// ---------------------------------------------------------------------------
__global__ void cvt_f16_k(const float* __restrict__ src,
                          _Float16* __restrict__ dst, long long n) {
  long long i = (long long)blockIdx.x * blockDim.x + threadIdx.x;
  if (i < n) dst[i] = (_Float16)src[i];
}

__global__ void fill_f32_k(float* __restrict__ p, long long n, float v) {
  long long i = (long long)blockIdx.x * blockDim.x + threadIdx.x;
  if (i < n) p[i] = v;
}

__global__ void deg_scatter_k(const int* __restrict__ dst,
                              float* __restrict__ deg, long long E) {
  long long e = (long long)blockIdx.x * blockDim.x + threadIdx.x;
  if (e < E) atomicAdd(&deg[dst[e]], 1.0f);
}

__global__ void rsqrt_inplace_k(float* __restrict__ p, long long n) {
  long long i = (long long)blockIdx.x * blockDim.x + threadIdx.x;
  if (i < n) p[i] = rsqrtf(p[i]);
}

// 32 lanes = one edge's 128 features (float4 per lane), coalesced L2 atomics.
__global__ void edge_scatter_k(const float* __restrict__ hw,
                               const int* __restrict__ src,
                               const int* __restrict__ dst,
                               const float* __restrict__ dinv,
                               float* __restrict__ out, long long E) {
  long long tid = (long long)blockIdx.x * blockDim.x + threadIdx.x;
  long long e = tid >> 5;
  if (e >= E) return;
  int fg = (int)(tid & 31) * 4;
  int s = src[e], d = dst[e];
  float nrm = dinv[s] * dinv[d];
  const float4 v = *(const float4*)(hw + (size_t)s * HC + fg);
  float* o = out + (size_t)d * HC + fg;
  atomicAdd(o + 0, v.x * nrm);
  atomicAdd(o + 1, v.y * nrm);
  atomicAdd(o + 2, v.z * nrm);
  atomicAdd(o + 3, v.w * nrm);
}

// Per-channel sum / sumsq over R rows (C = power of two).
__global__ void bn_stats_k(const float* __restrict__ x, float* __restrict__ sums,
                           long long R, int C) {
  int c = threadIdx.x & (C - 1);
  int rpb = blockDim.x / C;
  long long r = (long long)blockIdx.x * rpb + (threadIdx.x / C);
  long long stride = (long long)gridDim.x * rpb;
  float s = 0.0f, sq = 0.0f;
  for (; r < R; r += stride) {
    float v = x[r * C + c];
    s += v; sq += v * v;
  }
  atomicAdd(&sums[c], s);
  atomicAdd(&sums[C + c], sq);
}

// h_out = max(bn(x), 0) + res   (biased variance, training-mode BN)
__global__ void bn_apply_k(const float* __restrict__ x,
                           const float* __restrict__ sums,
                           const float* __restrict__ gamma,
                           const float* __restrict__ beta,
                           const float* __restrict__ res,
                           float* __restrict__ out,
                           long long n, int C, float invR) {
  long long i = (long long)blockIdx.x * blockDim.x + threadIdx.x;
  if (i >= n) return;
  int c = (int)(i & (C - 1));
  float m = sums[c] * invR;
  float v = sums[C + c] * invR - m * m;
  float y = (x[i] - m) * rsqrtf(v + EPSV) * gamma[c] + beta[c];
  out[i] = fmaxf(y, 0.0f) + res[i];
}

// Atomic mean-pool scatter: 32 lanes = one node's 128 features.
__global__ void pool_scatter_k(const float* __restrict__ h,
                               const int* __restrict__ batch,
                               float* __restrict__ pooled,
                               float* __restrict__ cnt, long long N) {
  long long tid = (long long)blockIdx.x * blockDim.x + threadIdx.x;
  long long nidx = tid >> 5;
  if (nidx >= N) return;
  int fg = (int)(tid & 31) * 4;
  int g = batch[nidx];
  const float4 v = *(const float4*)(h + (size_t)nidx * HC + fg);
  float* o = pooled + (size_t)g * HC + fg;
  atomicAdd(o + 0, v.x);
  atomicAdd(o + 1, v.y);
  atomicAdd(o + 2, v.z);
  atomicAdd(o + 3, v.w);
  if ((tid & 31) == 0) atomicAdd(&cnt[g], 1.0f);
}

__global__ void pool_div_k(float* __restrict__ pooled,
                           const float* __restrict__ cnt, long long n) {
  long long i = (long long)blockIdx.x * blockDim.x + threadIdx.x;
  if (i >= n) return;
  pooled[i] /= fmaxf(cnt[i >> 7], 1.0f);
}

// Fused: bn(z) -> relu -> dot with lin_w -> out[g].  One wave per graph row.
__global__ void final_head_k(const float* __restrict__ z,
                             const float* __restrict__ sums,
                             const float* __restrict__ fgam,
                             const float* __restrict__ fbet,
                             const float* __restrict__ lw,
                             const float* __restrict__ lb,
                             float* __restrict__ out, int G, float invR) {
  int g = blockIdx.x * (blockDim.x >> 5) + (threadIdx.x >> 5);
  if (g >= G) return;
  int lane = threadIdx.x & 31;
  float acc = 0.0f;
#pragma unroll
  for (int i = 0; i < 2; ++i) {
    int c = lane + i * 32;
    float m = sums[c] * invR;
    float v = sums[64 + c] * invR - m * m;
    float y = (z[(size_t)g * 64 + c] - m) * rsqrtf(v + EPSV) * fgam[c] + fbet[c];
    y = fmaxf(y, 0.0f);
    acc += y * lw[c];
  }
#pragma unroll
  for (int off = 16; off > 0; off >>= 1) acc += __shfl_xor(acc, off, 32);
  if (lane == 0) out[g] = acc + lb[0];
}

// ---------------------------------------------------------------------------
extern "C" void kernel_launch(void* const* d_in, const int* in_sizes, int n_in,
                              void* d_out, int out_size, void* d_ws, size_t ws_size,
                              hipStream_t stream) {
  (void)n_in; (void)ws_size;
  const float* x      = (const float*)d_in[0];
  const int*   ei     = (const int*)d_in[1];
  const int*   batch  = (const int*)d_in[2];
  const float* w_in   = (const float*)d_in[3];
  const float* pre_w  = (const float*)d_in[20];
  const float* pre_b  = (const float*)d_in[21];
  const float* fgam   = (const float*)d_in[22];
  const float* fbet   = (const float*)d_in[23];
  const float* lin_w  = (const float*)d_in[24];
  const float* lin_b  = (const float*)d_in[25];
  float* out = (float*)d_out;

  const long long N = in_sizes[2];
  const long long E = in_sizes[1] / 2;
  const long long G = out_size;
  const int*   src = ei;
  const int*   dst = ei + E;

  // workspace carving: every region 256-byte aligned (d_ws is 256-aligned)
  char* base = (char*)d_ws;
  size_t off = 0;
  auto carve = [&](size_t bytes) {
    void* p = base + off;
    off += (bytes + 255) & ~(size_t)255;
    return p;
  };
  float*     dinv     = (float*)carve(N * 4);
  float*     identity = (float*)carve(N * HC * 4);
  float*     h        = (float*)carve(N * HC * 4);
  float*     tmp1     = (float*)carve(N * HC * 4);   // gemm out (gather source)
  float*     tmp2     = (float*)carve(N * HC * 4);   // aggregation target
  float*     sums     = (float*)carve(256 * 4);
  float*     pooled   = (float*)carve(G * HC * 4);
  float*     cnt      = (float*)carve(G * 4);
  float*     zbuf     = (float*)carve(G * 64 * 4);
  _Float16*  w_in_h   = (_Float16*)carve(HC * HC * 2);
  _Float16*  conv_h0  = (_Float16*)carve(HC * HC * 2);
  _Float16*  conv_h1  = (_Float16*)carve(HC * HC * 2);
  _Float16*  conv_h2  = (_Float16*)carve(HC * HC * 2);
  _Float16*  conv_h3  = (_Float16*)carve(HC * HC * 2);
  _Float16*  pre_w_h  = (_Float16*)carve(64 * HC * 2);
  _Float16*  conv_h[4] = {conv_h0, conv_h1, conv_h2, conv_h3};

  const int TB = 256;
  auto blocks = [](long long n, int tb) { return (unsigned)((n + tb - 1) / tb); };

  // --- pre-convert weights to f16 (loop-invariant, L2-resident) ---
  cvt_f16_k<<<blocks(HC * HC, TB), TB, 0, stream>>>(w_in, w_in_h, HC * HC);
  for (int l = 0; l < 4; ++l)
    cvt_f16_k<<<blocks(HC * HC, TB), TB, 0, stream>>>(
        (const float*)d_in[4 + 4 * l], conv_h[l], HC * HC);
  cvt_f16_k<<<blocks(64 * HC, TB), TB, 0, stream>>>(pre_w, pre_w_h, 64 * HC);

  // --- degrees -> dinv (self-loop gives deg >= 1) ---
  fill_f32_k<<<blocks(N, TB), TB, 0, stream>>>(dinv, N, 1.0f);
  deg_scatter_k<<<blocks(E, TB), TB, 0, stream>>>(dst, dinv, E);
  rsqrt_inplace_k<<<blocks(N, TB), TB, 0, stream>>>(dinv, N);

  // --- identity = x @ w_in^T  (no bias, no fused epilogue) ---
  unsigned gemmN_blocks = blocks((N + 15) / 16, 8);   // 8 waves/block, 16 rows/wave
  wmma_gemm_k<8><<<gemmN_blocks, TB, 0, stream>>>(
      x, w_in_h, nullptr, identity, nullptr, nullptr, (int)N, HC);

  // --- 4 GCN layers ---
  const long long NE = N * HC;
  for (int layer = 0; layer < 4; ++layer) {
    const float* gl = (const float*)d_in[6 + 4 * layer];
    const float* bl = (const float*)d_in[7 + 4 * layer];
    const float* hin = (layer == 0) ? x : h;
    const float* res = (layer == 0) ? identity : h;

    // conv bias omitted: per-channel constant cancels exactly under BN mean.
    // Fused epilogue initializes tmp2 with the self-loop term hw * dinv^2.
    wmma_gemm_k<8><<<gemmN_blocks, TB, 0, stream>>>(
        hin, conv_h[layer], nullptr, tmp1, dinv, tmp2, (int)N, HC);
    edge_scatter_k<<<blocks(E * 32, TB), TB, 0, stream>>>(tmp1, src, dst, dinv,
                                                          tmp2, E);
    fill_f32_k<<<1, 256, 0, stream>>>(sums, 256, 0.0f);
    bn_stats_k<<<512, TB, 0, stream>>>(tmp2, sums, N, HC);
    bn_apply_k<<<blocks(NE, TB), TB, 0, stream>>>(tmp2, sums, gl, bl, res, h,
                                                  NE, HC, 1.0f / (float)N);
  }

  // --- global mean pool ---
  fill_f32_k<<<blocks(G * HC, TB), TB, 0, stream>>>(pooled, G * HC, 0.0f);
  fill_f32_k<<<blocks(G, TB), TB, 0, stream>>>(cnt, G, 0.0f);
  pool_scatter_k<<<blocks(N * 32, TB), TB, 0, stream>>>(h, batch, pooled, cnt, N);
  pool_div_k<<<blocks(G * HC, TB), TB, 0, stream>>>(pooled, cnt, G * HC);

  // --- head: z = pooled @ pre_w^T + pre_b ; bn ; relu ; dot lin_w + lin_b ---
  unsigned gemmG_blocks = blocks((G + 15) / 16, 8);
  wmma_gemm_k<4><<<gemmG_blocks, TB, 0, stream>>>(
      pooled, pre_w_h, pre_b, zbuf, nullptr, nullptr, (int)G, HC);
  fill_f32_k<<<1, 256, 0, stream>>>(sums, 128, 0.0f);
  bn_stats_k<<<32, TB, 0, stream>>>(zbuf, sums, G, 64);
  final_head_k<<<blocks(G * 32, TB), TB, 0, stream>>>(zbuf, sums, fgam, fbet,
                                                      lin_w, lin_b, out, (int)G,
                                                      1.0f / (float)G);
}
